// DynamicGRU_8933531976026
// MI455X (gfx1250) — matrix operand order repaired and checked by reading, not compiled, not run on hardware
//
#include <hip/hip_runtime.h>

// AUGRU (DIEN attention-gated GRU), fused single kernel for gfx1250.
// Transposed GEMM formulation: D[outcol, batch] = W^T[outcol,k] * [x;h]^T[k,batch]
// -> batch rides the WMMA N/lane axis: att/seqlen are per-lane scalars, output
// rows are contiguous b128 stores, and the recurrent h stays in C-layout
// registers; C-layout -> B-fragment conversion packs to f16 first, then does a
// single v_permlanex16 half-swap per packed dword.

typedef __attribute__((ext_vector_type(16))) _Float16 v16h;
typedef __attribute__((ext_vector_type(8)))  float    v8f;
typedef __attribute__((ext_vector_type(2)))  __fp16   fp16x2;

#define Bsz 2048
#define Tsz 200
#define Dsz 64
#define Hsz 64
#define WAVES 2
#define THREADS (WAVES * 32)
#define ROWS_PER_WAVE 16
#define NBLOCKS (Bsz / (WAVES * ROWS_PER_WAVE))   // 64 blocks

// fast activations: v_exp_f32 + v_rcp_f32, no IEEE div expansion
__device__ __forceinline__ float fsigmoid(float x) {
    return __builtin_amdgcn_rcpf(1.0f + __expf(-x));
}
__device__ __forceinline__ float ftanh(float x) {
    return 1.0f - 2.0f * __builtin_amdgcn_rcpf(__expf(2.0f * x) + 1.0f);
}

// v_cvt_pk_rtz_f16_f32: two f32 -> one packed-f16 dword
__device__ __forceinline__ unsigned pk2(float a, float b) {
    union { fp16x2 h; unsigned u; } c;
    c.h = __builtin_amdgcn_cvt_pkrtz(a, b);
    return c.u;
}

// xor-16 half-swap across the wave (pure VALU, no DS counter)
__device__ __forceinline__ unsigned lane_swap16(unsigned x) {
#if __has_builtin(__builtin_amdgcn_permlanex16)
    return (unsigned)__builtin_amdgcn_permlanex16(
        (int)x, (int)x, 0x76543210, 0xfedcba98, false, false);
#else
    return (unsigned)__builtin_amdgcn_ds_swizzle((int)x, 0x401F);
#endif
}

// 8 consecutive bias values for this lane's (outcol tile, half): 2x ds b128
__device__ __forceinline__ v8f load_bias8(const float* bp, int mt, int hi) {
    const float4* q = (const float4*)(bp + mt * 16 + hi * 8);
    float4 a = q[0], b = q[1];
    v8f c;
    c[0] = a.x; c[1] = a.y; c[2] = a.z; c[3] = a.w;
    c[4] = b.x; c[5] = b.y; c[6] = b.z; c[7] = b.w;
    return c;
}

// Two 16x16 C-layout f32 tiles (K-cols base..+15 / base+16..+31, batch on
// lanes) -> one 32x16 B-fragment (f16, K packed 2/VGPR).  Pack to f16 pairs
// first, then one permlanex16 per dword pair:
//   lo lane dword j   = own a0[j]        (K 2j,2j+1)
//   hi lane dword j   = partner's a1[j]  (K 16+2j,..)
//   lo lane dword j+4 = partner's a0[j]  (K 8+2j,..)
//   hi lane dword j+4 = own a1[j]        (K 24+2j,..)
__device__ __forceinline__ v16h make_bfrag(const float (&p0)[8],
                                           const float (&p1)[8], int hi) {
    union { unsigned u[8]; v16h v; } d;
#pragma unroll
    for (int j = 0; j < 4; ++j) {
        unsigned a0 = pk2(p0[2 * j], p0[2 * j + 1]);
        unsigned a1 = pk2(p1[2 * j], p1[2 * j + 1]);
        unsigned e  = hi ? a0 : a1;     // dword the partner half needs
        unsigned s  = lane_swap16(e);   // receive partner's dword
        d.u[j]     = hi ? s  : a0;
        d.u[j + 4] = hi ? a1 : s;
    }
    return d.v;
}

__global__ __launch_bounds__(THREADS)
__attribute__((amdgpu_waves_per_eu(8)))
void augru_fused(const float* __restrict__ xin,     // [B,T,D]
                 const int*   __restrict__ seqlen,  // [B]
                 const float* __restrict__ att,     // [B,T,1]
                 const float* __restrict__ gate_k,  // [128,128]
                 const float* __restrict__ gate_b,  // [128]
                 const float* __restrict__ cand_k,  // [128,64]
                 const float* __restrict__ cand_b,  // [64]
                 float*       __restrict__ out)     // [B,T,H]
{
    // W^T pre-packed in exact WMMA A-fragment order: frag(mt,kc) is
    // 32 lanes x 32B contiguous -> each lane reads its v16h with 2x ds b128.
    __shared__ __align__(32) _Float16 s_gfr[32 * 512]; // 8 mt x 4 kc
    __shared__ __align__(32) _Float16 s_cfr[16 * 512]; // 4 mt x 4 kc
    __shared__ float s_gb[128];
    __shared__ float s_cb[64];

    const int tid = threadIdx.x;

    // ---- one-time weight pack; this kernel's contiguous A k order (16-bit, 16x32 MxK):
    // lane<16: V0..V7 -> K 0..15 ; lane>=16: K 16..31.
    for (int e = tid; e < 32 * 512; e += THREADS) {
        int frag = e >> 9, lane = (e >> 4) & 31, idx = e & 15;
        int mt = frag >> 2, kc = frag & 3;
        int j = idx >> 1, w = idx & 1, half = lane >> 4;
        int k = kc * 32 + half * 16 + 2 * j + w;
        int m = mt * 16 + (lane & 15);
        s_gfr[e] = (_Float16)gate_k[k * (2 * Hsz) + m];
    }
    for (int e = tid; e < 16 * 512; e += THREADS) {
        int frag = e >> 9, lane = (e >> 4) & 31, idx = e & 15;
        int mt = frag >> 2, kc = frag & 3;
        int j = idx >> 1, w = idx & 1, half = lane >> 4;
        int k = kc * 32 + half * 16 + 2 * j + w;
        int m = mt * 16 + (lane & 15);
        s_cfr[e] = (_Float16)cand_k[k * Hsz + m];
    }
    for (int i = tid; i < 128; i += THREADS) s_gb[i] = gate_b[i];
    for (int i = tid; i < 64; i += THREADS)  s_cb[i] = cand_b[i];
    __syncthreads();

    const int wave = tid >> 5;
    const int lane = tid & 31;
    const int lo = lane & 15;     // batch column within tile
    const int hi = lane >> 4;     // which 8-row half of the tile
    const int bat = (blockIdx.x * WAVES + wave) * ROWS_PER_WAVE + lo;
    const int sl = seqlen[bat];

    const float* xrow = xin + (size_t)bat * Tsz * Dsz;
    const float* arow = att + (size_t)bat * Tsz;
    float*       orow = out + (size_t)bat * Tsz * Hsz;

    // Hidden state, C-layout: h[mt][v] = element (hcol mt*16+hi*8+v, batch lo)
    float h[4][8];
#pragma unroll
    for (int m = 0; m < 4; ++m)
#pragma unroll
        for (int v = 0; v < 8; ++v) h[m][v] = 0.0f;

#pragma unroll 1
    for (int t = 0; t < Tsz; ++t) {
        // ---- x -> B fragments (K x N, batch on lanes), straight from global.
        v16h xB0, xB1;
        {
            const float* p = xrow + t * Dsz + hi * 16;
#pragma unroll
            for (int i = 0; i < 16; ++i) xB0[i] = (_Float16)p[i];
            p += 32;
#pragma unroll
            for (int i = 0; i < 16; ++i) xB1[i] = (_Float16)p[i];
        }
        if (t + 1 < Tsz) {  // global_prefetch of next timestep's x
            __builtin_prefetch(xrow + (t + 1) * Dsz + hi * 16, 0, 1);
            __builtin_prefetch(xrow + (t + 1) * Dsz + hi * 16 + 32, 0, 1);
        }

        // ---- h -> B fragments (pack + permlanex16, register-only)
        v16h hB0 = make_bfrag(h[0], h[1], hi);
        v16h hB1 = make_bfrag(h[2], h[3], hi);

        // ---- r gates two tiles at a time; build rB fragment immediately
        v16h rB0, rB1;
        {
            float rh0[8], rh1[8];
#pragma unroll
            for (int pair = 0; pair < 2; ++pair) {
#pragma unroll
                for (int sub = 0; sub < 2; ++sub) {
                    int mt = pair * 2 + sub;
                    v8f acc = load_bias8(s_gb, mt, hi);
#pragma unroll
                    for (int kc = 0; kc < 4; ++kc) {
                        v16h a = *(const v16h*)&s_gfr[(mt * 4 + kc) * 512 + lane * 16];
                        v16h bf = (kc == 0) ? xB0 : (kc == 1) ? xB1
                                : (kc == 2) ? hB0 : hB1;
                        acc = __builtin_amdgcn_wmma_f32_16x16x32_f16(
                            false, a, false, bf, (short)0, acc, false, false);
                    }
                    float* rhp = sub ? rh1 : rh0;
#pragma unroll
                    for (int v = 0; v < 8; ++v)
                        rhp[v] = fsigmoid(acc[v]) * h[mt][v];
                }
                v16h f = make_bfrag(rh0, rh1, hi);
                if (pair == 0) rB0 = f; else rB1 = f;
            }
        }

        // ---- fused per-tile: u gate (tile mt+4) -> candidate (tile mt) -> blend
        const float at = arow[t];        // per-lane scalar (batch on lanes)
        const bool valid = (t < sl);
        float* obase = orow + t * Hsz;

#pragma unroll
        for (int mt = 0; mt < 4; ++mt) {
            v8f ua = load_bias8(s_gb, mt + 4, hi);
#pragma unroll
            for (int kc = 0; kc < 4; ++kc) {
                v16h a = *(const v16h*)&s_gfr[((mt + 4) * 4 + kc) * 512 + lane * 16];
                v16h bf = (kc == 0) ? xB0 : (kc == 1) ? xB1
                        : (kc == 2) ? hB0 : hB1;
                ua = __builtin_amdgcn_wmma_f32_16x16x32_f16(
                    false, a, false, bf, (short)0, ua, false, false);
            }
            v8f ca = load_bias8(s_cb, mt, hi);
#pragma unroll
            for (int kc = 0; kc < 4; ++kc) {
                v16h a = *(const v16h*)&s_cfr[(mt * 4 + kc) * 512 + lane * 16];
                v16h bf = (kc == 0) ? xB0 : (kc == 1) ? xB1
                        : (kc == 2) ? rB0 : rB1;
                ca = __builtin_amdgcn_wmma_f32_16x16x32_f16(
                    false, a, false, bf, (short)0, ca, false, false);
            }
            float o[8];
#pragma unroll
            for (int v = 0; v < 8; ++v) {
                float u  = fsigmoid(ua[v]);
                float c  = ftanh(ca[v]);
                float uh = (1.0f - at) * u;                // attention gate
                float nh = uh * h[mt][v] + (1.0f - uh) * c;
                h[mt][v] = valid ? nh : h[mt][v];           // freeze past seqlen
                o[v]     = valid ? nh : 0.0f;               // zero past seqlen
            }
            float4 s0 = make_float4(o[0], o[1], o[2], o[3]);
            float4 s1 = make_float4(o[4], o[5], o[6], o[7]);
            *(float4*)(obase + mt * 16 + hi * 8)     = s0;  // b128 stores
            *(float4*)(obase + mt * 16 + hi * 8 + 4) = s1;
        }
    }
}

extern "C" void kernel_launch(void* const* d_in, const int* in_sizes, int n_in,
                              void* d_out, int out_size, void* d_ws, size_t ws_size,
                              hipStream_t stream) {
    (void)in_sizes; (void)n_in; (void)out_size; (void)d_ws; (void)ws_size;
    const float* x  = (const float*)d_in[0];
    const int*   sl = (const int*)d_in[1];
    const float* a  = (const float*)d_in[2];
    const float* gk = (const float*)d_in[3];
    const float* gb = (const float*)d_in[4];
    const float* ck = (const float*)d_in[5];
    const float* cb = (const float*)d_in[6];
    float* o = (float*)d_out;
    augru_fused<<<NBLOCKS, THREADS, 0, stream>>>(x, sl, a, gk, gb, ck, cb, o);
}
